// STEncoderBlock_46866683134386
// MI455X (gfx1250) — matrix operand order repaired and checked by reading, not compiled, not run on hardware
//
#include <hip/hip_runtime.h>
#include <hip/hip_bf16.h>

typedef __bf16 bf16_t;
typedef __attribute__((ext_vector_type(16))) __bf16 v16bf;
typedef __attribute__((ext_vector_type(8)))  __bf16 v8bf;
typedef __attribute__((ext_vector_type(8)))  float  v8f;

#define TT 24
#define NN 512
#define DD 64
#define FF 256
#define BB 16
#define RTOT (BB*TT*NN)   /* 196608 rows */

__device__ __forceinline__ v8f wmma_bf16(v16bf a, v16bf b, v8f c) {
  // emits v_wmma_f32_16x16x32_bf16
  return __builtin_amdgcn_wmma_f32_16x16x32_bf16(false, a, false, b, (short)0, c,
                                                 false, false);
}

// A fragment: 16x32 (MxK) tile from row-major bf16, leading dim ld (elements).
// Lane m = lane&15 needs K = 8*half+0..7 and K = 16+8*half+0..7:
// two contiguous 16B runs per lane.
__device__ __forceinline__ v16bf load_fragA(const bf16_t* p, int ld) {
  int lane = threadIdx.x & 31;
  int half = lane >> 4, m = lane & 15;
  const bf16_t* r = p + (size_t)m * ld + half * 8;
  v8bf lo = *(const v8bf*)(r);
  v8bf hi = *(const v8bf*)(r + 16);
  return __builtin_shufflevector(lo, hi, 0, 1, 2, 3, 4, 5, 6, 7,
                                 8, 9, 10, 11, 12, 13, 14, 15);
}

// B fragment from a TRANSPOSED operand: Bt is N x K row-major, leading dim ld.
// Lane n = lane&15 needs Bt[n][8*half + 0..7] and Bt[n][16 + 8*half + 0..7]: two 16B runs.
__device__ __forceinline__ v16bf load_fragBT(const bf16_t* p, int ld) {
  int lane = threadIdx.x & 31;
  int half = lane >> 4, n = lane & 15;
  const bf16_t* r = p + (size_t)n * ld + half * 8;
  v8bf lo = *(const v8bf*)(r);
  v8bf hi = *(const v8bf*)(r + 16);
  return __builtin_shufflevector(lo, hi, 0, 1, 2, 3, 4, 5, 6, 7,
                                 8, 9, 10, 11, 12, 13, 14, 15);
}

// C/D fragment: 16x16 f32. Row = vgpr + 8*(lane>=16), col = lane&15.
__device__ __forceinline__ void store_fragC(float* p, int ld, v8f c, float bias) {
  int lane = threadIdx.x & 31;
  int half = lane >> 4, n = lane & 15;
#pragma unroll
  for (int vi = 0; vi < 8; ++vi)
    p[(size_t)(vi + 8 * half) * ld + n] = c[vi] + bias;
}

// Same but converting to bf16 (LDS staging for coalesced global stores)
__device__ __forceinline__ void store_fragC_bf(bf16_t* p, int ld, v8f c) {
  int lane = threadIdx.x & 31;
  int half = lane >> 4, n = lane & 15;
#pragma unroll
  for (int vi = 0; vi < 8; ++vi)
    p[(size_t)(vi + 8 * half) * ld + n] = (bf16_t)c[vi];
}

// ---------------------------------------------------------------- prep kernels
// Transpose-convert: src is batch x K x N f32 (row-major), dst is batch x N x K bf16.
__global__ void cvtT_kernel(const float* __restrict__ s, bf16_t* __restrict__ d,
                            int K, int N, int batch) {
  int i = blockIdx.x * blockDim.x + threadIdx.x;
  int tot = batch * K * N;
  if (i >= tot) return;
  int b = i / (K * N);
  int r = i - b * (K * N);
  int k = r / N, n = r % N;
  d[(size_t)b * K * N + (size_t)n * K + k] = (bf16_t)s[i];
}

__global__ __launch_bounds__(256)
void deg_kernel(const float* __restrict__ adj, float* __restrict__ dinv) {
  __shared__ float red[256];
  int m = blockIdx.x;
  float s = 0.f;
  for (int j = threadIdx.x; j < NN; j += 256) s += adj[(size_t)m * NN + j];
  red[threadIdx.x] = s;
  __syncthreads();
  for (int st = 128; st > 0; st >>= 1) {
    if ((int)threadIdx.x < st) red[threadIdx.x] += red[threadIdx.x + st];
    __syncthreads();
  }
  if (threadIdx.x == 0) {
    float deg = red[0] + 1.0f;                 // + identity diagonal
    dinv[m] = rsqrtf(fmaxf(deg, 1e-12f));      // clip(deg)^-0.5
  }
}

__global__ void ahat_kernel(const float* __restrict__ adj,
                            const float* __restrict__ dinv,
                            bf16_t* __restrict__ ahb) {
  int i = blockIdx.x * blockDim.x + threadIdx.x;
  if (i >= NN * NN) return;
  int m = i >> 9, n = i & (NN - 1);
  float a = adj[i] + (m == n ? 1.0f : 0.0f);
  ahb[i] = (bf16_t)(dinv[m] * a * dinv[n]);
}

// Tiled 32x32 bf16 transpose: src [BT][NN][DD] -> dstT [BT][DD][NN].
// Coalesced b128 on both global sides; shuffle through LDS.
__global__ __launch_bounds__(32)
void transpose_kernel(const bf16_t* __restrict__ src, bf16_t* __restrict__ dstT) {
  __shared__ __align__(16) bf16_t tile[32][33];
  int bid = blockIdx.x;
  const int TPB = (NN / 32) * (DD / 32);      // 32 tiles per bt slice
  int bt = bid / TPB;
  int rem = bid - bt * TPB;
  int n0 = (rem / (DD / 32)) * 32;
  int d0 = (rem % (DD / 32)) * 32;
  int l = threadIdx.x;

  const bf16_t* s = src + ((size_t)bt * NN + n0 + l) * DD + d0;
#pragma unroll
  for (int u = 0; u < 4; ++u) {
    v8bf vv = *(const v8bf*)(s + u * 8);
#pragma unroll
    for (int e = 0; e < 8; ++e) tile[l][u * 8 + e] = vv[e];
  }
  __syncthreads();
  bf16_t* dp = dstT + ((size_t)bt * DD + d0 + l) * NN + n0;
#pragma unroll
  for (int u = 0; u < 4; ++u) {
    v8bf vv;
#pragma unroll
    for (int e = 0; e < 8; ++e) vv[e] = tile[u * 8 + e][l];
    *(v8bf*)(dp + u * 8) = vv;
  }
}

// --------------------------------------------- fused temporal attention block
__global__ __launch_bounds__(32)
void attn_kernel(const float* __restrict__ x,
                 const bf16_t* __restrict__ wqT, const bf16_t* __restrict__ wkT,
                 const bf16_t* __restrict__ wvT, const bf16_t* __restrict__ woT,
                 const float* __restrict__ bq, const float* __restrict__ bk,
                 const float* __restrict__ bv, const float* __restrict__ bo,
                 const float* __restrict__ g_t, const float* __restrict__ b_t,
                 float* __restrict__ h, bf16_t* __restrict__ hb) {
  __shared__ __align__(16) bf16_t xb[32 * DD];
  __shared__ float  qs[32 * DD];
  __shared__ float  ks_[32 * DD];
  __shared__ float  vs[32 * DD];
  __shared__ __align__(16) bf16_t attb[32 * DD];
  __shared__ float  os[32 * DD];

  int lane = threadIdx.x;
  int bid = blockIdx.x;                 // b * NN + n
  int b = bid / NN, n = bid % NN;

  // stage x[b, 0..23, n, :] as bf16, pad rows 24..31 with zeros
  for (int idx = lane; idx < TT * DD; idx += 32) {
    int t = idx >> 6, d = idx & 63;
    xb[idx] = (bf16_t)x[(((size_t)b * TT + t) * NN + n) * DD + d];
  }
  for (int idx = TT * DD + lane; idx < 32 * DD; idx += 32) xb[idx] = (bf16_t)0.f;
  __syncthreads();

  // ---- Q,K,V projections (WMMA), processed sequentially to cap VGPR pressure.
  // NOTE: LDS destination selected via ternaries (an initializer-list array of
  // __shared__ pointers becomes an illegal addrspacecast static initializer).
#pragma unroll 1
  for (int w = 0; w < 3; ++w) {
    const bf16_t* W    = (w == 0) ? wqT : ((w == 1) ? wkT : wvT);
    const float*  bias = (w == 0) ? bq  : ((w == 1) ? bk  : bv);
    float*        O    = (w == 0) ? qs  : ((w == 1) ? ks_ : vs);
#pragma unroll
    for (int m = 0; m < 2; ++m) {
      v16bf a0 = load_fragA(xb + m * 16 * DD, DD);
      v16bf a1 = load_fragA(xb + m * 16 * DD + 32, DD);
#pragma unroll
      for (int nt = 0; nt < 4; ++nt) {
        v8f acc = {};
        acc = wmma_bf16(a0, load_fragBT(W + nt * 16 * DD, DD), acc);
        acc = wmma_bf16(a1, load_fragBT(W + nt * 16 * DD + 32, DD), acc);
        store_fragC(O + m * 16 * DD + nt * 16, DD, acc, bias[nt * 16 + (lane & 15)]);
      }
    }
  }
  __syncthreads();

  // ---- softmax attention in f32 VALU (T=24, hd=16; ~2% of FLOPs)
  for (int it = lane; it < TT * 4; it += 32) {
    int t = it >> 2, hh = it & 3;
    const float* qr = qs + t * DD + hh * 16;
    float sc[TT];
    float mx = -1e30f;
#pragma unroll
    for (int j = 0; j < TT; ++j) {
      const float* kr = ks_ + j * DD + hh * 16;
      float s = 0.f;
#pragma unroll
      for (int d = 0; d < 16; ++d) s += qr[d] * kr[d];
      s *= 0.25f;                       // hd^-0.5
      sc[j] = s;
      mx = fmaxf(mx, s);
    }
    float den = 0.f;
#pragma unroll
    for (int j = 0; j < TT; ++j) { sc[j] = __expf(sc[j] - mx); den += sc[j]; }
    float inv = 1.0f / den;
    float out[16];
#pragma unroll
    for (int d = 0; d < 16; ++d) out[d] = 0.f;
#pragma unroll
    for (int j = 0; j < TT; ++j) {
      float w = sc[j] * inv;
      const float* vr = vs + j * DD + hh * 16;
#pragma unroll
      for (int d = 0; d < 16; ++d) out[d] += w * vr[d];
    }
#pragma unroll
    for (int d = 0; d < 16; ++d) attb[t * DD + hh * 16 + d] = (bf16_t)out[d];
  }
  for (int idx = TT * DD + lane; idx < 32 * DD; idx += 32) attb[idx] = (bf16_t)0.f;
  __syncthreads();

  // ---- output projection att @ Wo (WMMA)
#pragma unroll
  for (int m = 0; m < 2; ++m) {
    v16bf a0 = load_fragA(attb + m * 16 * DD, DD);
    v16bf a1 = load_fragA(attb + m * 16 * DD + 32, DD);
#pragma unroll
    for (int nt = 0; nt < 4; ++nt) {
      int col = nt * 16 + (lane & 15);
      v8f acc = {};
      acc = wmma_bf16(a0, load_fragBT(woT + nt * 16 * DD, DD), acc);
      acc = wmma_bf16(a1, load_fragBT(woT + nt * 16 * DD + 32, DD), acc);
      store_fragC(os + m * 16 * DD + nt * 16, DD, acc, bo[col]);
    }
  }
  __syncthreads();

  // ---- residual + LayerNorm -> h (f32), hb (bf16 row-major)
  if (lane < TT) {
    int t = lane;
    size_t row = ((size_t)b * TT + t) * NN + n;
    const float* xr = x + row * DD;
    float buf[DD];
    float s = 0.f, ss = 0.f;
#pragma unroll
    for (int d = 0; d < DD; ++d) {
      float y = xr[d] + os[t * DD + d];
      buf[d] = y; s += y; ss += y * y;
    }
    float mean = s * (1.0f / DD);
    float var = ss * (1.0f / DD) - mean * mean;
    float inv = rsqrtf(var + 1e-5f);
#pragma unroll
    for (int d = 0; d < DD; ++d) {
      float y = (buf[d] - mean) * inv * g_t[d] + b_t[d];
      h[row * DD + d] = y;
      hb[row * DD + d] = (bf16_t)y;
    }
  }
}

// ------------------------------------------ graph propagation: dst = Ahat@src
// srcT is the D-major (transposed) activation slice; dst row-major bf16,
// written coalesced via an LDS-staged tile.
__global__ __launch_bounds__(32)
void gprop_kernel(const bf16_t* __restrict__ ahb, const bf16_t* __restrict__ srcT,
                  bf16_t* __restrict__ dst) {
  __shared__ __align__(16) bf16_t ot[32 * DD];
  int bid = blockIdx.x;
  int bt = bid >> 4;                    // which (b,t) slice
  int m0 = (bid & 15) * 32;             // node-row tile
  int lane = threadIdx.x;

  v8f acc[2][4] = {};
  for (int kb = 0; kb < 16; ++kb) {
    int k0 = kb * 32;
    v16bf a0 = load_fragA(ahb + (size_t)m0 * NN + k0, NN);
    v16bf a1 = load_fragA(ahb + (size_t)(m0 + 16) * NN + k0, NN);
#pragma unroll
    for (int nt = 0; nt < 4; ++nt) {
      v16bf bf = load_fragBT(srcT + ((size_t)bt * DD + nt * 16) * NN + k0, NN);
      acc[0][nt] = wmma_bf16(a0, bf, acc[0][nt]);
      acc[1][nt] = wmma_bf16(a1, bf, acc[1][nt]);
    }
  }
#pragma unroll
  for (int i = 0; i < 2; ++i)
#pragma unroll
    for (int nt = 0; nt < 4; ++nt)
      store_fragC_bf(ot + i * 16 * DD + nt * 16, DD, acc[i][nt]);
  __syncthreads();

  // coalesced copy-out: lane r streams row r (64 bf16 = 8 x b128)
  {
    const v8bf* s = (const v8bf*)(ot + lane * DD);
    v8bf* d = (v8bf*)(dst + ((size_t)bt * NN + m0 + lane) * DD);
#pragma unroll
    for (int u = 0; u < 8; ++u) d[u] = s[u];
  }
}

// -------------------- fused K-hop mixing + LN + FFN(GELU) + LN -> final output
__global__ __launch_bounds__(32)
void gcnffn_kernel(const float* __restrict__ h, const bf16_t* __restrict__ hb,
                   const bf16_t* __restrict__ p1b, const bf16_t* __restrict__ p2b,
                   const bf16_t* __restrict__ gcwT, const float* __restrict__ gcb,
                   const float* __restrict__ g_g, const float* __restrict__ b_g,
                   const bf16_t* __restrict__ w1T, const float* __restrict__ b1,
                   const bf16_t* __restrict__ w2T, const float* __restrict__ b2,
                   const float* __restrict__ g_f, const float* __restrict__ b_f,
                   float* __restrict__ out) {
  __shared__ float  smf[32 * DD];       // gout, later FFN output
  __shared__ float  hgf[32 * DD];       // post-graph LN (f32)
  __shared__ __align__(16) bf16_t hgb[32 * DD];   // post-graph LN (bf16)
  __shared__ __align__(16) bf16_t midb[32 * FF];  // GELU(hg@W1+b1) (bf16)

  int lane = threadIdx.x;
  int half = lane >> 4;
  size_t row0 = (size_t)blockIdx.x * 32;

  // ---- gout = h@gcW0 + p1@gcW1 + p2@gcW2  (+ sum of gcb biases)
  {
    v8f acc[2][4] = {};
#pragma unroll 1
    for (int kk = 0; kk < 3; ++kk) {
      const bf16_t* S = (kk == 0) ? hb : ((kk == 1) ? p1b : p2b);
      const bf16_t* W = gcwT + kk * DD * DD;
#pragma unroll
      for (int i = 0; i < 2; ++i) {
        v16bf a0 = load_fragA(S + (row0 + i * 16) * DD, DD);
        v16bf a1 = load_fragA(S + (row0 + i * 16) * DD + 32, DD);
#pragma unroll
        for (int nt = 0; nt < 4; ++nt) {
          acc[i][nt] = wmma_bf16(a0, load_fragBT(W + nt * 16 * DD, DD), acc[i][nt]);
          acc[i][nt] = wmma_bf16(a1, load_fragBT(W + nt * 16 * DD + 32, DD), acc[i][nt]);
        }
      }
    }
#pragma unroll
    for (int i = 0; i < 2; ++i)
#pragma unroll
      for (int nt = 0; nt < 4; ++nt) {
        int col = nt * 16 + (lane & 15);
        float bias = gcb[col] + gcb[DD + col] + gcb[2 * DD + col];
        store_fragC(smf + i * 16 * DD + nt * 16, DD, acc[i][nt], bias);
      }
  }
  __syncthreads();

  // ---- hg = LN(h + gout)
  {
    int t = lane;
    size_t row = row0 + t;
    const float* hr = h + row * DD;
    float buf[DD];
    float s = 0.f, ss = 0.f;
#pragma unroll
    for (int d = 0; d < DD; ++d) {
      float y = hr[d] + smf[t * DD + d];
      buf[d] = y; s += y; ss += y * y;
    }
    float mean = s * (1.0f / DD);
    float var = ss * (1.0f / DD) - mean * mean;
    float inv = rsqrtf(var + 1e-5f);
#pragma unroll
    for (int d = 0; d < DD; ++d) {
      float y = (buf[d] - mean) * inv * g_g[d] + b_g[d];
      hgf[t * DD + d] = y;
      hgb[t * DD + d] = (bf16_t)y;
    }
  }
  __syncthreads();

  // ---- FFN layer 1: mid = GELU(hg @ W1 + b1), GELU fused into frag epilogue
#pragma unroll
  for (int i = 0; i < 2; ++i) {
    v16bf a0 = load_fragA(hgb + i * 16 * DD, DD);
    v16bf a1 = load_fragA(hgb + i * 16 * DD + 32, DD);
#pragma unroll 4
    for (int nt = 0; nt < 16; ++nt) {
      v8f acc = {};
      acc = wmma_bf16(a0, load_fragBT(w1T + nt * 16 * DD, DD), acc);
      acc = wmma_bf16(a1, load_fragBT(w1T + nt * 16 * DD + 32, DD), acc);
      int col = nt * 16 + (lane & 15);
      float bias = b1[col];
#pragma unroll
      for (int vi = 0; vi < 8; ++vi) {
        float vx = acc[vi] + bias;
        vx = 0.5f * vx * (1.0f + erff(vx * 0.70710678118654752f)); // exact GELU
        midb[(size_t)(i * 16 + vi + 8 * half) * FF + col] = (bf16_t)vx;
      }
    }
  }
  __syncthreads();

  // ---- FFN layer 2: f = mid @ W2 + b2
#pragma unroll
  for (int i = 0; i < 2; ++i) {
    v16bf am[8];
#pragma unroll
    for (int kb = 0; kb < 8; ++kb)
      am[kb] = load_fragA(midb + (size_t)i * 16 * FF + kb * 32, FF);
#pragma unroll
    for (int nt = 0; nt < 4; ++nt) {
      v8f acc = {};
#pragma unroll
      for (int kb = 0; kb < 8; ++kb)
        acc = wmma_bf16(am[kb], load_fragBT(w2T + (size_t)nt * 16 * FF + kb * 32, FF), acc);
      int col = nt * 16 + (lane & 15);
      store_fragC(smf + i * 16 * DD + nt * 16, DD, acc, b2[col]);
    }
  }
  __syncthreads();

  // ---- out = LN(hg + f)
  {
    int t = lane;
    size_t row = row0 + t;
    float buf[DD];
    float s = 0.f, ss = 0.f;
#pragma unroll
    for (int d = 0; d < DD; ++d) {
      float y = hgf[t * DD + d] + smf[t * DD + d];
      buf[d] = y; s += y; ss += y * y;
    }
    float mean = s * (1.0f / DD);
    float var = ss * (1.0f / DD) - mean * mean;
    float inv = rsqrtf(var + 1e-5f);
#pragma unroll
    for (int d = 0; d < DD; ++d)
      out[row * DD + d] = (buf[d] - mean) * inv * g_f[d] + b_f[d];
  }
}

// ------------------------------------------------------------------- launcher
extern "C" void kernel_launch(void* const* d_in, const int* in_sizes, int n_in,
                              void* d_out, int out_size, void* d_ws, size_t ws_size,
                              hipStream_t stream) {
  (void)in_sizes; (void)n_in; (void)out_size; (void)ws_size;
  const float* x   = (const float*)d_in[0];
  const float* adj = (const float*)d_in[1];
  const float* Wq  = (const float*)d_in[2];
  const float* bq  = (const float*)d_in[3];
  const float* Wk  = (const float*)d_in[4];
  const float* bk  = (const float*)d_in[5];
  const float* Wv  = (const float*)d_in[6];
  const float* bv  = (const float*)d_in[7];
  const float* Wo  = (const float*)d_in[8];
  const float* bo  = (const float*)d_in[9];
  const float* gcW = (const float*)d_in[10];
  const float* gcb = (const float*)d_in[11];
  const float* W1  = (const float*)d_in[12];
  const float* b1  = (const float*)d_in[13];
  const float* W2  = (const float*)d_in[14];
  const float* b2  = (const float*)d_in[15];
  const float* g_t = (const float*)d_in[16];
  const float* b_t = (const float*)d_in[17];
  const float* g_g = (const float*)d_in[18];
  const float* b_g = (const float*)d_in[19];
  const float* g_f = (const float*)d_in[20];
  const float* b_f = (const float*)d_in[21];
  float* out = (float*)d_out;

  char* ws = (char*)d_ws;
  size_t off = 0;
  auto take = [&](size_t bytes) -> char* {
    char* p = ws + off;
    off += (bytes + 255) & ~(size_t)255;
    return p;
  };
  float*  dinv = (float*)take(NN * sizeof(float));
  bf16_t* ahb  = (bf16_t*)take((size_t)NN * NN * sizeof(bf16_t));
  bf16_t* wqT  = (bf16_t*)take((size_t)DD * DD * sizeof(bf16_t));
  bf16_t* wkT  = (bf16_t*)take((size_t)DD * DD * sizeof(bf16_t));
  bf16_t* wvT  = (bf16_t*)take((size_t)DD * DD * sizeof(bf16_t));
  bf16_t* woT  = (bf16_t*)take((size_t)DD * DD * sizeof(bf16_t));
  bf16_t* gcwT = (bf16_t*)take((size_t)3 * DD * DD * sizeof(bf16_t));
  bf16_t* w1T  = (bf16_t*)take((size_t)DD * FF * sizeof(bf16_t));
  bf16_t* w2T  = (bf16_t*)take((size_t)FF * DD * sizeof(bf16_t));
  float*  hf   = (float*)take((size_t)RTOT * DD * sizeof(float));
  bf16_t* hb   = (bf16_t*)take((size_t)RTOT * DD * sizeof(bf16_t));
  bf16_t* hbT  = (bf16_t*)take((size_t)RTOT * DD * sizeof(bf16_t));
  bf16_t* p1b  = (bf16_t*)take((size_t)RTOT * DD * sizeof(bf16_t));
  bf16_t* p1bT = (bf16_t*)take((size_t)RTOT * DD * sizeof(bf16_t));
  bf16_t* p2b  = (bf16_t*)take((size_t)RTOT * DD * sizeof(bf16_t));

  // transposed bf16 weight staging (every WMMA B-fragment = two 16B runs per lane)
  cvtT_kernel<<<dim3((DD * DD + 255) / 256), dim3(256), 0, stream>>>(Wq, wqT, DD, DD, 1);
  cvtT_kernel<<<dim3((DD * DD + 255) / 256), dim3(256), 0, stream>>>(Wk, wkT, DD, DD, 1);
  cvtT_kernel<<<dim3((DD * DD + 255) / 256), dim3(256), 0, stream>>>(Wv, wvT, DD, DD, 1);
  cvtT_kernel<<<dim3((DD * DD + 255) / 256), dim3(256), 0, stream>>>(Wo, woT, DD, DD, 1);
  cvtT_kernel<<<dim3((3 * DD * DD + 255) / 256), dim3(256), 0, stream>>>(gcW, gcwT, DD, DD, 3);
  cvtT_kernel<<<dim3((DD * FF + 255) / 256), dim3(256), 0, stream>>>(W1, w1T, DD, FF, 1);
  cvtT_kernel<<<dim3((FF * DD + 255) / 256), dim3(256), 0, stream>>>(W2, w2T, FF, DD, 1);

  // normalized adjacency (bf16, 0.5MB -> stays resident in L2 for both hops)
  deg_kernel<<<dim3(NN), dim3(256), 0, stream>>>(adj, dinv);
  ahat_kernel<<<dim3((NN * NN + 255) / 256), dim3(256), 0, stream>>>(adj, dinv, ahb);

  // fused temporal attention + LN  -> h (f32) + hb (bf16)
  attn_kernel<<<dim3(BB * NN), dim3(32), 0, stream>>>(
      x, wqT, wkT, wvT, woT, bq, bk, bv, bo, g_t, b_t, hf, hb);

  const int TGRID = BB * TT * (NN / 32) * (DD / 32);   // transpose tiles
  // h^T for hop-1 B-operand
  transpose_kernel<<<dim3(TGRID), dim3(32), 0, stream>>>(hb, hbT);
  // p1 = Ahat @ h ; p1^T ; p2 = Ahat @ p1
  gprop_kernel<<<dim3(BB * TT * (NN / 32)), dim3(32), 0, stream>>>(ahb, hbT, p1b);
  transpose_kernel<<<dim3(TGRID), dim3(32), 0, stream>>>(p1b, p1bT);
  gprop_kernel<<<dim3(BB * TT * (NN / 32)), dim3(32), 0, stream>>>(ahb, p1bT, p2b);

  // graph mixing + LN + FFN + LN -> out
  gcnffn_kernel<<<dim3(RTOT / 32), dim3(32), 0, stream>>>(
      hf, hb, p1b, p2b, gcwT, gcb, g_g, b_g, w1T, b1, w2T, b2, g_f, b_f, out);
}
